// DCN_85134841741350
// MI455X (gfx1250) — matrix operand order repaired and checked by reading, not compiled, not run on hardware
//
#include <hip/hip_runtime.h>

typedef __attribute__((ext_vector_type(16))) _Float16 v16h;
typedef __attribute__((ext_vector_type(8)))  _Float16 v8h;
typedef __attribute__((ext_vector_type(8)))  float    v8f;

#define N_IMG   4
#define C_IN    64
#define H_IMG   128
#define W_IMG   128
#define HW      (H_IMG * W_IMG)          // 16384
#define COUT    64
#define KK      9
#define CK      (C_IN * KK)              // 576
#define OMCH    27                        // 18 offset + 9 mask channels
#define SPITCH  592                       // padded f16 LDS row pitch (16B aligned)
#define PXB     64                        // pixels per block

// LDS byte budget:
//   sA   : 64 * SPITCH f16  = 75776 B   (w_om rows 0..31 first, then w_dcn rows 0..63)
//   sB   : 64 * SPITCH f16  = 75776 B   (im2col patch first, then sampled values)
//   sOM  : 32 * 64 f32      =  8192 B   (offset/mask tile, channels 0..26 used)
//   sWt  : 576 * 4 f32      =  9216 B
//   sIdx : 576 * 4 i32      =  9216 B
//   total = 178176 B  (needs CDNA5's 320KB WGP LDS)
#define SMEM_BYTES (2 * 64 * SPITCH * 2 + 32 * 64 * 4 + (KK * PXB) * 4 * 4 * 2)

__global__ __launch_bounds__(256, 1)
void dcn_fused_wmma(const float* __restrict__ x,
                    const float* __restrict__ w_om,
                    const float* __restrict__ b_om,
                    const float* __restrict__ w_dcn,
                    float* __restrict__ out)
{
    extern __shared__ __align__(16) char smem[];
    _Float16* sA   = (_Float16*)smem;                 // 64*SPITCH
    _Float16* sB   = sA + 64 * SPITCH;                // 64*SPITCH
    float*    sOM  = (float*)(sB + 64 * SPITCH);      // 32*64
    float*    sWt  = sOM + 32 * PXB;                  // 576*4
    int*      sIdx = (int*)(sWt + (KK * PXB) * 4);    // 576*4

    const int tid = threadIdx.x;
    const int blk = blockIdx.x;
    const int n   = blk >> 8;                 // 256 tiles per image (HW/PXB)
    const int p0  = (blk & 255) * PXB;        // pixel base within image (one row: W=128)
    const int oy  = p0 >> 7;
    const int ox0 = p0 & (W_IMG - 1);

    const float* xn = x + (size_t)n * C_IN * HW;

    const int wave  = tid >> 5;
    const int lane  = tid & 31;
    const int r     = lane & 15;
    const int khalf = (lane < 16) ? 0 : 16;  // A frag: contiguous K half per lane
    const int bk    = (lane < 16) ? 0 : 16;  // B frag: contiguous K half per lane
    const int m0    = (lane < 16) ? 0 : 8;   // C/D row base per lane half

    // ---- Phase 1a: w_om -> sA rows 0..31 (zero-pad rows 27..31) ----
    for (int i = tid; i < 32 * CK; i += 256) {
        const int row = i / CK;
        const int col = i - row * CK;
        sA[row * SPITCH + col] = (row < OMCH) ? (_Float16)w_om[row * CK + col]
                                              : (_Float16)0.0f;
    }
    // ---- Phase 1b: im2col patch -> sB[px][ck], ck = c*9 + kh*3 + kw ----
    for (int t = tid; t < CK * PXB; t += 256) {
        const int px = t & (PXB - 1);
        const int ck = t >> 6;
        const int c  = ck / KK;
        const int k  = ck - c * KK;
        const int kh = k / 3;
        const int kw = k - kh * 3;
        const int iy = oy - 1 + kh;
        const int ix = ox0 + px - 1 + kw;
        float v = 0.0f;
        if (iy >= 0 && iy < H_IMG && ix >= 0 && ix < W_IMG)
            v = xn[c * HW + iy * W_IMG + ix];
        sB[px * SPITCH + ck] = (_Float16)v;
    }
    __syncthreads();

    // ---- Phase 2: offset/mask GEMM via WMMA. 8 tiles (2 oc x 4 px), 1 per wave ----
    {
        const int ocT = wave >> 2;           // 0..1  (rows 0..15 / 16..31)
        const int pxT = wave & 3;            // 0..3
        v8f acc = {0.f, 0.f, 0.f, 0.f, 0.f, 0.f, 0.f, 0.f};
        const _Float16* arow = sA + (ocT * 16 + r) * SPITCH;
        const _Float16* brow = sB + (pxT * 16 + r) * SPITCH;
        for (int kk = 0; kk < CK; kk += 32) {
            const v16h a = *(const v16h*)(arow + kk + khalf);
            const v16h b = *(const v16h*)(brow + kk + bk);
            acc = __builtin_amdgcn_wmma_f32_16x16x32_f16(
                false, a, false, b, (short)0, acc, false, false);
        }
        const int px = pxT * 16 + r;
#pragma unroll
        for (int g = 0; g < 8; ++g) {
            const int ch = ocT * 16 + m0 + g;
            if (ch < OMCH) {
                const float v = acc[g] + b_om[ch];
                sOM[ch * PXB + px] = (ch < 18) ? v : (1.0f / (1.0f + __expf(-v)));
            }
        }
    }
    __syncthreads();

    // ---- Phase 3a: w_dcn -> sA (64 rows, f16) ----
    for (int i = tid; i < COUT * CK; i += 256) {
        const int row = i / CK;
        const int col = i - row * CK;
        sA[row * SPITCH + col] = (_Float16)w_dcn[i];
    }
    // ---- Phase 3b: bilinear positions from LDS offmask. e = k*64 + px ----
    for (int e = tid; e < KK * PXB; e += 256) {
        const int k  = e >> 6;
        const int px = e & (PXB - 1);
        const int ox = ox0 + px;

        const float offy = sOM[(2 * k) * PXB + px];
        const float offx = sOM[(2 * k + 1) * PXB + px];
        const float m    = sOM[(18 + k) * PXB + px];

        const float yy = (float)(oy - 1 + k / 3) + offy;
        const float xx = (float)(ox - 1 + k % 3) + offx;
        const float y0f = floorf(yy), x0f = floorf(xx);
        const float wy1 = yy - y0f,  wx1 = xx - x0f;
        const int   y0  = (int)y0f,  x0  = (int)x0f;

        const float cw[4] = { (1.f - wy1) * (1.f - wx1), (1.f - wy1) * wx1,
                              wy1 * (1.f - wx1),          wy1 * wx1 };
#pragma unroll
        for (int j = 0; j < 4; ++j) {
            const int  yc = y0 + (j >> 1);
            const int  xc = x0 + (j & 1);
            const bool valid = (yc >= 0) && (yc < H_IMG) && (xc >= 0) && (xc < W_IMG);
            const int  yi = min(max(yc, 0), H_IMG - 1);
            const int  xi = min(max(xc, 0), W_IMG - 1);
            sIdx[e * 4 + j] = yi * W_IMG + xi;
            sWt [e * 4 + j] = valid ? cw[j] * m : 0.0f;
        }
    }
    __syncthreads();

    // ---- Phase 4: masked bilinear sampling -> sB[px][ck] (L2-served gathers) ----
    for (int t = tid; t < C_IN * KK * PXB; t += 256) {
        const int px = t & (PXB - 1);
        const int ck = t >> 6;
        const int c  = ck / KK;
        const int k  = ck - c * KK;
        const int e  = k * PXB + px;
        const float* xc = xn + c * HW;
        const float v = sWt[e * 4 + 0] * xc[sIdx[e * 4 + 0]]
                      + sWt[e * 4 + 1] * xc[sIdx[e * 4 + 1]]
                      + sWt[e * 4 + 2] * xc[sIdx[e * 4 + 2]]
                      + sWt[e * 4 + 3] * xc[sIdx[e * 4 + 3]];
        sB[px * SPITCH + ck] = (_Float16)v;
    }
    __syncthreads();

    // ---- Phase 5: main GEMM. 16 tiles (4 oc x 4 px) over 8 waves, K=576 ----
    for (int t2 = wave; t2 < 16; t2 += 8) {
        const int ocT = t2 >> 2;
        const int pxT = t2 & 3;
        v8f acc = {0.f, 0.f, 0.f, 0.f, 0.f, 0.f, 0.f, 0.f};

        const _Float16* arow = sA + (ocT * 16 + r) * SPITCH;
        const _Float16* brow = sB + (pxT * 16 + r) * SPITCH;

        for (int kk = 0; kk < CK; kk += 32) {
            const v16h a = *(const v16h*)(arow + kk + khalf);
            const v16h b = *(const v16h*)(brow + kk + bk);
            acc = __builtin_amdgcn_wmma_f32_16x16x32_f16(
                false, a, false, b, (short)0, acc, false, false);
        }

        float* op = out + ((size_t)n * COUT + ocT * 16 + m0) * HW + p0 + r;
#pragma unroll
        for (int g = 0; g < 8; ++g) op[g * HW] = acc[g];
    }
}

// ---------------------------------------------------------------------------
extern "C" void kernel_launch(void* const* d_in, const int* in_sizes, int n_in,
                              void* d_out, int out_size, void* d_ws, size_t ws_size,
                              hipStream_t stream)
{
    (void)in_sizes; (void)n_in; (void)out_size; (void)d_ws; (void)ws_size;
    const float* x     = (const float*)d_in[0];
    const float* w_om  = (const float*)d_in[1];
    const float* b_om  = (const float*)d_in[2];
    const float* w_dcn = (const float*)d_in[3];
    float*       out   = (float*)d_out;

    hipFuncSetAttribute((const void*)dcn_fused_wmma,
                        hipFuncAttributeMaxDynamicSharedMemorySize, (int)SMEM_BYTES);
    dcn_fused_wmma<<<(N_IMG * HW) / PXB, 256, SMEM_BYTES, stream>>>(
        x, w_om, b_om, w_dcn, out);
}
